// SetupConditionedBandFusion_17008070492584
// MI455X (gfx1250) — matrix operand order repaired and checked by reading, not compiled, not run on hardware
//
#include <hip/hip_runtime.h>
#include <cstdint>
#include <cstddef>

// ---------------- constants ----------------
constexpr int   kB   = 32;
constexpr int   kT   = 2048;
constexpr int   kR   = 256;
constexpr int   kCD  = 512;   // 2*R
constexpr int   kSD  = 128;
constexpr float kEPS = 1e-5f;

// ---------------- types ----------------
typedef __attribute__((ext_vector_type(16))) __bf16 v16bf;
typedef __attribute__((ext_vector_type(8)))  float  v8f;
typedef __attribute__((ext_vector_type(4)))  float  fx4;

struct __align__(16) US8 { unsigned short u[8]; };
union  FragBF { v16bf v; US8 q[2]; };

__device__ __forceinline__ unsigned short f2bf(float x) {
  unsigned int u = __float_as_uint(x);
  u += 0x7FFFu + ((u >> 16) & 1u);          // round-to-nearest-even
  return (unsigned short)(u >> 16);
}
__device__ __forceinline__ float geluf(float x) {
  return 0.5f * x * (1.0f + erff(x * 0.70710678118654752f));
}

// ---------------- utility kernels ----------------
__global__ void k_zero(float* __restrict__ p, int n) {
  int i = blockIdx.x * 256 + threadIdx.x;
  if (i < n) p[i] = 0.f;
}

__global__ void k_f32_to_bf16(const float* __restrict__ in,
                              unsigned short* __restrict__ out, int n) {
  int i = blockIdx.x * 256 + threadIdx.x;
  if (i < n) out[i] = f2bf(in[i]);
}

// denom[b] = max(sum_t mask[b,t], 1)
__global__ void k_denom(const int* __restrict__ mask, float* __restrict__ denom) {
  __shared__ float rs[256];
  const int b = blockIdx.x, tid = threadIdx.x;
  float s = 0.f;
  for (int t = tid; t < kT; t += 256) s += (float)mask[b * kT + t];
  rs[tid] = s; __syncthreads();
  for (int st = 128; st > 0; st >>= 1) {
    if (tid < st) rs[tid] += rs[tid + st];
    __syncthreads();
  }
  if (tid == 0) denom[b] = fmaxf(rs[0], 1.f);
}

// masked column sums of one band: pooled[f][b][c] += sum over a T-chunk
// block = 128 threads; each thread owns 4 contiguous columns (one b128 load/row)
__global__ void k_pool(const float* __restrict__ band, const int* __restrict__ mask,
                       float* __restrict__ pooled, int f) {
  const int b = blockIdx.x, chunk = blockIdx.y, tid = threadIdx.x;
  const int col = tid * 4;
  fx4 acc = {0.f, 0.f, 0.f, 0.f};
  const int tb = chunk * 256, te = tb + 256;
  for (int t = tb; t < te; ++t) {
    if (mask[b * kT + t]) {            // wave-uniform: skip masked-out rows
      const size_t base = ((size_t)b * kT + t) * kCD;
      acc += *(const fx4*)(band + base + col);
    }
  }
  float* pr = pooled + ((size_t)f * kB + b) * kCD + col;
  atomicAdd(pr + 0, acc[0]);
  atomicAdd(pr + 1, acc[1]);
  atomicAdd(pr + 2, acc[2]);
  atomicAdd(pr + 3, acc[3]);
}

// tokens[b][0..2] = LN(pooled/denom; sn), tokens[b][3] = z_setup @ sp_w^T + sp_b
__global__ void k_tokens(const float* __restrict__ pooled, const float* __restrict__ denom,
                         const float* __restrict__ sn_g, const float* __restrict__ sn_b,
                         const float* __restrict__ zsetup, const float* __restrict__ sp_w,
                         const float* __restrict__ sp_b,
                         float* __restrict__ tokens, unsigned short* __restrict__ tok_bf) {
  __shared__ float rs[256], rs2[256];
  const int b = blockIdx.x, tid = threadIdx.x;
  const float inv_dn = 1.f / denom[b];
  for (int f = 0; f < 3; ++f) {
    const float* p = pooled + ((size_t)f * kB + b) * kCD;
    float s = 0.f, s2 = 0.f;
    for (int c = tid; c < kCD; c += 256) {
      const float v = p[c] * inv_dn; s += v; s2 += v * v;
    }
    rs[tid] = s; rs2[tid] = s2; __syncthreads();
    for (int st = 128; st > 0; st >>= 1) {
      if (tid < st) { rs[tid] += rs[tid + st]; rs2[tid] += rs2[tid + st]; }
      __syncthreads();
    }
    const float mu = rs[0] * (1.f / kCD);
    const float var = rs2[0] * (1.f / kCD) - mu * mu;
    const float r = rsqrtf(var + kEPS);
    for (int c = tid; c < kCD; c += 256) {
      const float v = p[c] * inv_dn;
      const float y = (v - mu) * r * sn_g[c] + sn_b[c];
      const size_t o = ((size_t)b * 4 + f) * kCD + c;
      tokens[o] = y; tok_bf[o] = f2bf(y);
    }
    __syncthreads();
  }
  for (int c = tid; c < kCD; c += 256) {
    float s = sp_b[c];
    const float* wr = sp_w + (size_t)c * kSD;
    const float* z  = zsetup + b * kSD;
    for (int j = 0; j < kSD; ++j) s += wr[j] * z[j];
    const size_t o = ((size_t)b * 4 + 3) * kCD + c;
    tokens[o] = s; tok_bf[o] = f2bf(s);
  }
}

// ---------------- generic bf16 WMMA GEMM: C[M,N] = A[M,K] @ W[N,K]^T + bias ----------------
// block = 128 threads (4 waves); grid = (N/64, M/16); act: 0=none, 1=gelu
__global__ void k_gemm_bf16(const unsigned short* __restrict__ A,
                            const unsigned short* __restrict__ W,
                            const float* __restrict__ bias,
                            float* __restrict__ C,
                            int M, int N, int K, int act) {
  const int wv    = threadIdx.x >> 5;
  const int lane  = threadIdx.x & 31;
  const int ntile = blockIdx.x * 4 + wv;
  const int mtile = blockIdx.y;
  const int hf    = lane >> 4;     // half-wave select
  const int l16   = lane & 15;
  const unsigned short* arow = A + (size_t)(mtile * 16 + l16) * K;
  const unsigned short* wrow = W + (size_t)(ntile * 16 + l16) * K;
  v8f acc = {0.f, 0.f, 0.f, 0.f, 0.f, 0.f, 0.f, 0.f};
  for (int k0 = 0; k0 < K; k0 += 64) {      // K always a multiple of 64 here
    FragBF a0, b0, a1, b1;
    a0.q[0] = *(const US8*)(arow + k0 + hf * 8);
    a0.q[1] = *(const US8*)(arow + k0 + 16 + hf * 8);
    b0.q[0] = *(const US8*)(wrow + k0 + hf * 8);
    b0.q[1] = *(const US8*)(wrow + k0 + 16 + hf * 8);
    a1.q[0] = *(const US8*)(arow + k0 + 32 + hf * 8);
    a1.q[1] = *(const US8*)(arow + k0 + 48 + hf * 8);
    b1.q[0] = *(const US8*)(wrow + k0 + 32 + hf * 8);
    b1.q[1] = *(const US8*)(wrow + k0 + 48 + hf * 8);
    acc = __builtin_amdgcn_wmma_f32_16x16x32_bf16(false, a0.v, false, b0.v,
                                                  (short)0, acc, false, false);
    acc = __builtin_amdgcn_wmma_f32_16x16x32_bf16(false, a1.v, false, b1.v,
                                                  (short)0, acc, false, false);
  }
  const int coln = ntile * 16 + l16;
  const float bv = bias ? bias[coln] : 0.f;
#pragma unroll
  for (int i = 0; i < 8; ++i) {
    const int row = mtile * 16 + hf * 8 + i;   // D layout: VGPR i -> M = i + 8*half
    float v = acc[i] + bv;
    if (act == 1) v = geluf(v);
    C[(size_t)row * N + coln] = v;
  }
}

// ---------------- tiny 4-token attention (per batch) ----------------
__global__ void k_attn(const float* __restrict__ qkv, float* __restrict__ ao) {
  __shared__ float att[4][4][4];  // [h][sq][sk]
  const int b = blockIdx.x, tid = threadIdx.x;  // 128 threads
  if (tid < 64) {
    const int sq = tid >> 4, h = (tid >> 2) & 3, sk = tid & 3;
    const float* q = qkv + ((size_t)(b * 4 + sq)) * 1536 + h * 128;
    const float* k = qkv + ((size_t)(b * 4 + sk)) * 1536 + 512 + h * 128;
    float s = 0.f;
    for (int d = 0; d < 128; ++d) s += q[d] * k[d];
    att[h][sq][sk] = s * 0.08838834764831845f;  // 1/sqrt(128)
  }
  __syncthreads();
  if (tid < 16) {
    const int h = tid >> 2, sq = tid & 3;
    float mx = att[h][sq][0];
    for (int j = 1; j < 4; ++j) mx = fmaxf(mx, att[h][sq][j]);
    float e[4], sum = 0.f;
    for (int j = 0; j < 4; ++j) { e[j] = expf(att[h][sq][j] - mx); sum += e[j]; }
    const float inv = 1.f / sum;
    for (int j = 0; j < 4; ++j) att[h][sq][j] = e[j] * inv;
  }
  __syncthreads();
  for (int j = tid; j < 4 * kCD; j += 128) {
    const int sq = j >> 9, d = j & 511, h = d >> 7;
    float s = 0.f;
    for (int sk = 0; sk < 4; ++sk)
      s += att[h][sq][sk] * qkv[((size_t)(b * 4 + sk)) * 1536 + 1024 + d];
    ao[((size_t)(b * 4 + sq)) * kCD + d] = s;
  }
}

// out = LN(A[row] + B[row]; g, be); optional f32 and bf16 outputs
__global__ void k_addln(const float* __restrict__ A, const float* __restrict__ Bv,
                        const float* __restrict__ g, const float* __restrict__ be,
                        float* __restrict__ outF, unsigned short* __restrict__ outBF,
                        int D) {
  __shared__ float rs[256], rs2[256];
  const int row = blockIdx.x, tid = threadIdx.x;
  const float* a  = A + (size_t)row * D;
  const float* bb = Bv ? Bv + (size_t)row * D : nullptr;
  float s = 0.f, s2 = 0.f;
  for (int c = tid; c < D; c += 256) {
    const float v = a[c] + (bb ? bb[c] : 0.f); s += v; s2 += v * v;
  }
  rs[tid] = s; rs2[tid] = s2; __syncthreads();
  for (int st = 128; st > 0; st >>= 1) {
    if (tid < st) { rs[tid] += rs[tid + st]; rs2[tid] += rs2[tid + st]; }
    __syncthreads();
  }
  const float mu = rs[0] / D, var = rs2[0] / D - mu * mu, r = rsqrtf(var + kEPS);
  for (int c = tid; c < D; c += 256) {
    const float v = a[c] + (bb ? bb[c] : 0.f);
    const float y = (v - mu) * r * g[c] + be[c];
    if (outF)  outF[(size_t)row * D + c]  = y;
    if (outBF) outBF[(size_t)row * D + c] = f2bf(y);
  }
}

// Build LN'd [refined_band | z_setup] rows (bf16) and band-major refined bf16 rows.
// grid = 96 blocks: row = band*32 + b
__global__ void k_scat(const float* __restrict__ refined, const float* __restrict__ zsetup,
                       const float* __restrict__ g, const float* __restrict__ be,
                       unsigned short* __restrict__ scat_bf, unsigned short* __restrict__ A96) {
  __shared__ float rs[256], rs2[256];
  const int row = blockIdx.x, band = row >> 5, b = row & 31, tid = threadIdx.x;
  const int D = kCD + kSD;  // 640
  const float* rrow = refined + ((size_t)b * 4 + band) * kCD;
  const float* z    = zsetup + b * kSD;
  float s = 0.f, s2 = 0.f;
  for (int c = tid; c < D; c += 256) {
    const float v = (c < kCD) ? rrow[c] : z[c - kCD]; s += v; s2 += v * v;
  }
  rs[tid] = s; rs2[tid] = s2; __syncthreads();
  for (int st = 128; st > 0; st >>= 1) {
    if (tid < st) { rs[tid] += rs[tid + st]; rs2[tid] += rs2[tid + st]; }
    __syncthreads();
  }
  const float mu = rs[0] / D, var = rs2[0] / D - mu * mu, r = rsqrtf(var + kEPS);
  for (int c = tid; c < D; c += 256) {
    const float v = (c < kCD) ? rrow[c] : z[c - kCD];
    scat_bf[(size_t)row * D + c] = f2bf((v - mu) * r * g[c] + be[c]);
  }
  for (int c = tid; c < kCD; c += 256) A96[(size_t)row * kCD + c] = f2bf(rrow[c]);
}

// scores + band softmax -> bw (ws copy + output copy)
__global__ void k_score(const float* __restrict__ smh, const float* __restrict__ sm2_w,
                        const float* __restrict__ sm2_b,
                        float* __restrict__ bw_ws, float* __restrict__ out_bw) {
  __shared__ float sc[96];
  const int tid = threadIdx.x;  // 128
  if (tid < 96) {
    const float* hr = smh + (size_t)tid * 256;
    float s = sm2_b[0];
    for (int j = 0; j < 256; ++j) s += hr[j] * sm2_w[j];
    sc[tid] = s;
  }
  __syncthreads();
  if (tid < 32) {
    const int b = tid;
    const float a0 = sc[b], a1 = sc[32 + b], a2 = sc[64 + b];
    const float mx = fmaxf(a0, fmaxf(a1, a2));
    const float e0 = expf(a0 - mx), e1 = expf(a1 - mx), e2 = expf(a2 - mx);
    const float inv = 1.f / (e0 + e1 + e2);
    const float w0 = e0 * inv, w1 = e1 * inv, w2 = e2 * inv;
    bw_ws[b * 3 + 0] = w0; bw_ws[b * 3 + 1] = w1; bw_ws[b * 3 + 2] = w2;
    out_bw[b * 3 + 0] = w0; out_bw[b * 3 + 1] = w1; out_bw[b * 3 + 2] = w2;
  }
}

// ---------------- the big streaming pass ----------------
// h = w0*b0 + w1*b1 + w2*b2 (NT store), y = LN(h; cn), ypool[b] += mask ? y : 0
// grid = (T/64, B), block = 256 (8 waves, one row per wave at a time).
// Each lane owns 16 fixed columns -> accumulate normalized values in registers,
// apply gamma/beta once at flush (sum of per-row beta = nkept*beta), single LDS
// atomic flush per lane, single global atomic flush per column per block.
__global__ void k_fuse(const float* __restrict__ b0, const float* __restrict__ b1,
                       const float* __restrict__ b2, const int* __restrict__ mask,
                       const float* __restrict__ bw, const float* __restrict__ cn_g,
                       const float* __restrict__ cn_b, float* __restrict__ outH,
                       float* __restrict__ ypool) {
  __shared__ float ypart[kCD];
  const int b = blockIdx.y, t0 = blockIdx.x * 64, tid = threadIdx.x;
  for (int c = tid; c < kCD; c += 256) ypart[c] = 0.f;
  __syncthreads();
  const float w0 = bw[b * 3 + 0], w1 = bw[b * 3 + 1], w2 = bw[b * 3 + 2];
  const int wv = tid >> 5, lane = tid & 31;
  float yacc[16];
#pragma unroll
  for (int i = 0; i < 16; ++i) yacc[i] = 0.f;
  float nk = 0.f;
  for (int r = wv; r < 64; r += 8) {
    const int t = t0 + r;
    const size_t base = ((size_t)b * kT + t) * kCD;
    fx4 h4[4];
#pragma unroll
    for (int j = 0; j < 4; ++j) {
      const int col = j * 128 + lane * 4;
      const fx4 a0 = __builtin_nontemporal_load((const fx4*)(b0 + base + col));
      const fx4 a1 = __builtin_nontemporal_load((const fx4*)(b1 + base + col));
      const fx4 a2 = __builtin_nontemporal_load((const fx4*)(b2 + base + col));
      h4[j] = w0 * a0 + w1 * a1 + w2 * a2;
      __builtin_nontemporal_store(h4[j], (fx4*)(outH + base + col));
    }
    if (mask[b * kT + t]) {          // wave-uniform branch: LN only for kept rows
      float s = 0.f, s2 = 0.f;
#pragma unroll
      for (int j = 0; j < 4; ++j)
#pragma unroll
        for (int q = 0; q < 4; ++q) { const float v = h4[j][q]; s += v; s2 += v * v; }
#pragma unroll
      for (int d = 1; d < 32; d <<= 1) {
        s  += __shfl_xor(s,  d, 32);
        s2 += __shfl_xor(s2, d, 32);
      }
      const float mu   = s  * (1.f / kCD);
      const float var  = s2 * (1.f / kCD) - mu * mu;
      const float rstd = rsqrtf(var + kEPS);
      nk += 1.f;
#pragma unroll
      for (int j = 0; j < 4; ++j)
#pragma unroll
        for (int q = 0; q < 4; ++q) yacc[j * 4 + q] += (h4[j][q] - mu) * rstd;
    }
  }
#pragma unroll
  for (int j = 0; j < 4; ++j)
#pragma unroll
    for (int q = 0; q < 4; ++q) {
      const int col = j * 128 + lane * 4 + q;
      atomicAdd(&ypart[col], yacc[j * 4 + q] * cn_g[col] + nk * cn_b[col]);
    }
  __syncthreads();
  for (int c = tid; c < kCD; c += 256) atomicAdd(&ypool[b * kCD + c], ypart[c]);
}

__global__ void k_ypoolbf(const float* __restrict__ ypool, const float* __restrict__ denom,
                          unsigned short* __restrict__ obf) {
  const int b = blockIdx.x, tid = threadIdx.x;
  const float inv = 1.f / denom[b];
  for (int c = tid; c < kCD; c += 256) obf[b * kCD + c] = f2bf(ypool[b * kCD + c] * inv);
}

// ---------------- launcher ----------------
extern "C" void kernel_launch(void* const* d_in, const int* in_sizes, int n_in,
                              void* d_out, int out_size, void* d_ws, size_t ws_size,
                              hipStream_t stream) {
  const float* band0  = (const float*)d_in[0];
  const float* band1  = (const float*)d_in[1];
  const float* band2  = (const float*)d_in[2];
  const float* zsetup = (const float*)d_in[3];
  const int*   mask   = (const int*)  d_in[4];
  const float* sn_g   = (const float*)d_in[5];
  const float* sn_b   = (const float*)d_in[6];
  const float* sp_w   = (const float*)d_in[7];
  const float* sp_b   = (const float*)d_in[8];
  const float* in_w   = (const float*)d_in[9];
  const float* in_b   = (const float*)d_in[10];
  const float* out_w  = (const float*)d_in[11];
  const float* out_b  = (const float*)d_in[12];
  const float* n1_g   = (const float*)d_in[13];
  const float* n1_b   = (const float*)d_in[14];
  const float* f1_w   = (const float*)d_in[15];
  const float* f1_b   = (const float*)d_in[16];
  const float* f2_w   = (const float*)d_in[17];
  const float* f2_b   = (const float*)d_in[18];
  const float* n2_g   = (const float*)d_in[19];
  const float* n2_b   = (const float*)d_in[20];
  const float* smln_g = (const float*)d_in[21];
  const float* smln_b = (const float*)d_in[22];
  const float* sm1_w  = (const float*)d_in[23];
  const float* sm1_b  = (const float*)d_in[24];
  const float* sm2_w  = (const float*)d_in[25];
  const float* sm2_b  = (const float*)d_in[26];
  const float* b2r_w  = (const float*)d_in[27];
  const float* b2r_b  = (const float*)d_in[28];
  const float* cn_g   = (const float*)d_in[29];
  const float* cn_b   = (const float*)d_in[30];
  const float* c2r_w  = (const float*)d_in[31];
  const float* c2r_b  = (const float*)d_in[32];

  float* out = (float*)d_out;
  const size_t H_OFF  = 0;
  const size_t CG_OFF = (size_t)kB * kT * kCD;           // 33,554,432
  const size_t BR_OFF = CG_OFF + (size_t)kB * kR;        // +8,192
  const size_t BW_OFF = BR_OFF + (size_t)3 * kB * kR;    // +24,576

  // -------- workspace bump allocator --------
  char* wsp = (char*)d_ws; size_t off = 0;
  auto alloc = [&](size_t bytes) -> void* {
    void* p = wsp + off; off += (bytes + 255) & ~(size_t)255; return p;
  };
  float* denom   = (float*)alloc(kB * 4);
  float* pooled  = (float*)alloc((size_t)3 * kB * kCD * 4);
  float* tokens  = (float*)alloc((size_t)kB * 4 * kCD * 4);
  float* qkv     = (float*)alloc((size_t)kB * 4 * 3 * kCD * 4);
  float* ao      = (float*)alloc((size_t)kB * 4 * kCD * 4);
  float* proj    = (float*)alloc((size_t)kB * 4 * kCD * 4);
  float* t1      = (float*)alloc((size_t)kB * 4 * kCD * 4);
  float* ff1     = (float*)alloc((size_t)kB * 4 * 2 * kCD * 4);
  float* ff2     = (float*)alloc((size_t)kB * 4 * kCD * 4);
  float* refined = (float*)alloc((size_t)kB * 4 * kCD * 4);
  float* smh     = (float*)alloc((size_t)96 * 256 * 4);
  float* bw_ws   = (float*)alloc((size_t)kB * 3 * 4);
  float* ypool   = (float*)alloc((size_t)kB * kCD * 4);
  unsigned short* tok_bf   = (unsigned short*)alloc((size_t)kB * 4 * kCD * 2);
  unsigned short* ao_bf    = (unsigned short*)alloc((size_t)kB * 4 * kCD * 2);
  unsigned short* t1_bf    = (unsigned short*)alloc((size_t)kB * 4 * kCD * 2);
  unsigned short* ff1_bf   = (unsigned short*)alloc((size_t)kB * 4 * 2 * kCD * 2);
  unsigned short* scat_bf  = (unsigned short*)alloc((size_t)96 * (kCD + kSD) * 2);
  unsigned short* A96      = (unsigned short*)alloc((size_t)96 * kCD * 2);
  unsigned short* yp_bf    = (unsigned short*)alloc((size_t)kB * kCD * 2);
  unsigned short* in_wb    = (unsigned short*)alloc((size_t)3 * kCD * kCD * 2);
  unsigned short* out_wb   = (unsigned short*)alloc((size_t)kCD * kCD * 2);
  unsigned short* f1_wb    = (unsigned short*)alloc((size_t)2 * kCD * kCD * 2);
  unsigned short* f2_wb    = (unsigned short*)alloc((size_t)kCD * 2 * kCD * 2);
  unsigned short* sm1_wb   = (unsigned short*)alloc((size_t)256 * (kCD + kSD) * 2);
  unsigned short* b2r_wb   = (unsigned short*)alloc((size_t)kR * kCD * 2);
  unsigned short* c2r_wb   = (unsigned short*)alloc((size_t)kR * kCD * 2);

  // -------- stage 0: pools + denom --------
  k_zero<<<dim3((3 * kB * kCD + 255) / 256), 256, 0, stream>>>(pooled, 3 * kB * kCD);
  k_zero<<<dim3((kB * kCD + 255) / 256), 256, 0, stream>>>(ypool, kB * kCD);
  k_denom<<<dim3(kB), 256, 0, stream>>>(mask, denom);
  k_pool<<<dim3(kB, 8), 128, 0, stream>>>(band0, mask, pooled, 0);
  k_pool<<<dim3(kB, 8), 128, 0, stream>>>(band1, mask, pooled, 1);
  k_pool<<<dim3(kB, 8), 128, 0, stream>>>(band2, mask, pooled, 2);

  // -------- weight -> bf16 conversions (L2-resident, reused) --------
  auto cvt = [&](const float* src, unsigned short* dst, int n) {
    k_f32_to_bf16<<<dim3((n + 255) / 256), 256, 0, stream>>>(src, dst, n);
  };
  cvt(in_w,  in_wb,  3 * kCD * kCD);
  cvt(out_w, out_wb, kCD * kCD);
  cvt(f1_w,  f1_wb,  2 * kCD * kCD);
  cvt(f2_w,  f2_wb,  kCD * 2 * kCD);
  cvt(sm1_w, sm1_wb, 256 * (kCD + kSD));
  cvt(b2r_w, b2r_wb, kR * kCD);
  cvt(c2r_w, c2r_wb, kR * kCD);

  // -------- stage 1: tokens, 4-token transformer (WMMA GEMMs) --------
  k_tokens<<<dim3(kB), 256, 0, stream>>>(pooled, denom, sn_g, sn_b,
                                         zsetup, sp_w, sp_b, tokens, tok_bf);
  // qkv = tokens @ in_w^T : [128,1536]
  k_gemm_bf16<<<dim3(1536 / 64, 128 / 16), 128, 0, stream>>>(
      tok_bf, in_wb, in_b, qkv, 128, 1536, 512, 0);
  k_attn<<<dim3(kB), 128, 0, stream>>>(qkv, ao);
  cvt(ao, ao_bf, 128 * kCD);
  k_gemm_bf16<<<dim3(512 / 64, 128 / 16), 128, 0, stream>>>(
      ao_bf, out_wb, out_b, proj, 128, 512, 512, 0);
  k_addln<<<dim3(128), 256, 0, stream>>>(tokens, proj, n1_g, n1_b, t1, t1_bf, kCD);
  k_gemm_bf16<<<dim3(1024 / 64, 128 / 16), 128, 0, stream>>>(
      t1_bf, f1_wb, f1_b, ff1, 128, 1024, 512, 1 /*gelu*/);
  cvt(ff1, ff1_bf, 128 * 1024);
  k_gemm_bf16<<<dim3(512 / 64, 128 / 16), 128, 0, stream>>>(
      ff1_bf, f2_wb, f2_b, ff2, 128, 512, 1024, 0);
  k_addln<<<dim3(128), 256, 0, stream>>>(t1, ff2, n2_g, n2_b, refined, nullptr, kCD);

  // -------- stage 2: band scores -> bw, band_rel --------
  k_scat<<<dim3(96), 256, 0, stream>>>(refined, zsetup, smln_g, smln_b, scat_bf, A96);
  k_gemm_bf16<<<dim3(256 / 64, 96 / 16), 128, 0, stream>>>(
      scat_bf, sm1_wb, sm1_b, smh, 96, 256, 640, 1 /*gelu*/);
  k_score<<<dim3(1), 128, 0, stream>>>(smh, sm2_w, sm2_b, bw_ws, out + BW_OFF);
  // band_rel: rows are band-major (band*32+b) -> exactly the output layout
  k_gemm_bf16<<<dim3(256 / 64, 96 / 16), 128, 0, stream>>>(
      A96, b2r_wb, b2r_b, out + BR_OFF, 96, 256, 512, 0);

  // -------- stage 3: fused band mix + LN + masked y-pool (HBM-bound) --------
  k_fuse<<<dim3(kT / 64, kB), 256, 0, stream>>>(band0, band1, band2, mask, bw_ws,
                                                cn_g, cn_b, out + H_OFF, ypool);

  // -------- stage 4: cg = pool(LN(h)) @ c2r_w^T + c2r_b  (GEMM collapsed by linearity) --------
  k_ypoolbf<<<dim3(kB), 256, 0, stream>>>(ypool, denom, yp_bf);
  k_gemm_bf16<<<dim3(256 / 64, 32 / 16), 128, 0, stream>>>(
      yp_bf, c2r_wb, c2r_b, out + CG_OFF, 32, 256, 512, 0);

  (void)in_sizes; (void)n_in; (void)out_size; (void)ws_size;
}
